// ModeONet_2147483648612
// MI455X (gfx1250) — matrix operand order repaired and checked by reading, not compiled, not run on hardware
//
#include <hip/hip_runtime.h>

typedef __attribute__((ext_vector_type(16))) _Float16 v16h;
typedef __attribute__((ext_vector_type(8)))  float    v8f;

#define BB   8
#define FF   128
#define PP   32
#define NWH  9216            // 96*96
#define NCHUNKS 72           // 9216 / 128
#define LN_ELEMS   9437184   // 8*128*9216
#define AUX_OFF    28311552

// ---------------- workspace layout (floats) ----------------
// w_inv   : 256        inverse norms per (b,p)
// w_bias  : 1024       bias per (b,f)
// w_invf  : 1024       1/fnorm per (b,f)
// w_aux   : 256        aux partials per (b,p)
// w_fp    : 73728      fnorm partials [b][ftile][nchunk][16]
// w_tr/ti : 32768 f16  each, scaled t matrices

__device__ __forceinline__ float block_reduce_256(float v, float* sm) {
    #pragma unroll
    for (int off = 16; off > 0; off >>= 1) v += __shfl_down(v, off);
    if ((threadIdx.x & 31) == 0) sm[threadIdx.x >> 5] = v;
    __syncthreads();
    float t = 0.f;
    if (threadIdx.x == 0) {
        #pragma unroll
        for (int i = 0; i < 8; ++i) t += sm[i];
    }
    return t;
}

// Shared tile engine: build A (16x32) / B (32x16) f16 fragments with a contiguous
// k order shared by A and B and run the 4-WMMA complex MMA.
//   A: lane holds row M=m; element i -> K = 16*h + i
//   B: lane holds col n;  element i -> K = 16*h + i
__device__ __forceinline__ void complex_tile(const float* __restrict__ shapes,
                                             const _Float16* __restrict__ w_tr,
                                             const _Float16* __restrict__ w_ti,
                                             const float* s_inv,
                                             int b, int fbase, int n, int m, int h,
                                             v8f& acc_r, v8f& acc_i) {
    const _Float16* pr = w_tr + (size_t)(b * FF + fbase + m) * PP;
    const _Float16* pi = w_ti + (size_t)(b * FF + fbase + m) * PP;
    v16h a_r, a_i, a_in;
    #pragma unroll
    for (int i = 0; i < 16; ++i) {
        a_r[i] = pr[16 * h + i];
        a_i[i] = pi[16 * h + i];
    }
    #pragma unroll
    for (int i = 0; i < 16; ++i) a_in[i] = (_Float16)(-a_i[i]);

    const float* sr = shapes + (size_t)(b * 2 + 0) * PP * NWH + n;
    const float* si = shapes + (size_t)(b * 2 + 1) * PP * NWH + n;
    v16h b_r, b_i;
    #pragma unroll
    for (int i = 0; i < 16; ++i) {
        const int K = 16 * h + i;
        const float sc = s_inv[K];
        b_r[i] = (_Float16)(sr[(size_t)K * NWH] * sc);
        b_i[i] = (_Float16)(si[(size_t)K * NWH] * sc);
    }

    acc_r = v8f{};
    acc_i = v8f{};
    acc_r = __builtin_amdgcn_wmma_f32_16x16x32_f16(false, a_r,  false, b_r, (short)0, acc_r, false, false);
    acc_r = __builtin_amdgcn_wmma_f32_16x16x32_f16(false, a_in, false, b_i, (short)0, acc_r, false, false);
    acc_i = __builtin_amdgcn_wmma_f32_16x16x32_f16(false, a_r,  false, b_i, (short)0, acc_i, false, false);
    acc_i = __builtin_amdgcn_wmma_f32_16x16x32_f16(false, a_i,  false, b_r, (short)0, acc_i, false, false);
}

// 1) inverse norms: one block per (b,p)
__global__ __launch_bounds__(256) void k_norm(const float* __restrict__ shapes,
                                              float* __restrict__ w_inv) {
    __shared__ float sm[8];
    const int bp = blockIdx.x, b = bp >> 5, p = bp & 31;
    const float* s0 = shapes + ((size_t)(b * 2 + 0) * PP + p) * NWH;
    const float* s1 = shapes + ((size_t)(b * 2 + 1) * PP + p) * NWH;
    float sum = 0.f;
    for (int j = threadIdx.x; j < NWH; j += 256) {
        float a = s0[j], c = s1[j];
        sum += a * a + c * c;
    }
    sum = block_reduce_256(sum, sm);
    if (threadIdx.x == 0) w_inv[bp] = 1.0f / (sqrtf(sum) + 1e-8f);
}

// 2) bias interp + f16 conversion of t/P : one thread per (b,f)
__global__ __launch_bounds__(256) void k_prep(const float* __restrict__ resp,
                                              const float* __restrict__ freqs,
                                              const float* __restrict__ table,
                                              float* __restrict__ w_bias,
                                              _Float16* __restrict__ w_tr,
                                              _Float16* __restrict__ w_ti) {
    const int t = blockIdx.x * 256 + threadIdx.x;
    if (t >= BB * FF) return;
    float x = freqs[t];
    float idx = fminf(fmaxf((x + 1.0f) * 0.5f * 249.0f, 0.0f), 249.0f);
    int   i0 = (int)floorf(idx);
    int   i1 = min(i0 + 1, 249);
    float wgt = idx - (float)i0;
    w_bias[t] = table[i0] * (1.0f - wgt) + table[i1] * wgt;

    const float* rp = resp + (size_t)t * 2 * PP;       // [2][P]
    _Float16* trp = w_tr + (size_t)t * PP;
    _Float16* tip = w_ti + (size_t)t * PP;
    #pragma unroll
    for (int p = 0; p < PP; ++p) {
        trp[p] = (_Float16)(rp[p]      * (1.0f / (float)PP));
        tip[p] = (_Float16)(rp[PP + p] * (1.0f / (float)PP));
    }
}

// 3) aux partials: one block per (b,p), 256 threads = F*2 elements
__global__ __launch_bounds__(256) void k_aux(const float* __restrict__ resp,
                                             float* __restrict__ w_aux) {
    __shared__ float sm[8];
    const int bp = blockIdx.x, b = bp >> 5, p = bp & 31;
    const int f = threadIdx.x >> 1, c = threadIdx.x & 1;
    float x = resp[((size_t)(b * FF + f) * 2 + c) * PP + p];
    float sum = block_reduce_256(x * x, sm);
    if (threadIdx.x == 0) w_aux[bp] = logf(sqrtf(sum) + 1.0f);
}

// 4) GEMM pass 1: ln_mag + deterministic fnorm partials (no field write).
//    One wave per 16x16 (f x n) tile, 8 waves/block; grid = b * 8 ftiles * 72 nchunks
__global__ __launch_bounds__(256) void k_gemm_reduce(const float* __restrict__ shapes,
                                                     const _Float16* __restrict__ w_tr,
                                                     const _Float16* __restrict__ w_ti,
                                                     const float* __restrict__ w_inv,
                                                     const float* __restrict__ w_bias,
                                                     float* __restrict__ out_ln,
                                                     float* __restrict__ w_fp) {
    const int bx = blockIdx.x;
    const int nchunk = bx % NCHUNKS;
    const int ft = (bx / NCHUNKS) & 7;
    const int b  = bx / (NCHUNKS * 8);
    const int tid = threadIdx.x;
    const int w = tid >> 5;
    const int lane = tid & 31;
    const int m = lane & 15;
    const int h = lane >> 4;

    __shared__ float s_inv[32];
    __shared__ float s_bias[16];
    __shared__ float s_rows[8][16];
    if (tid < 32) s_inv[tid] = w_inv[b * 32 + tid];
    if (tid < 16) s_bias[tid] = w_bias[b * FF + ft * 16 + tid];
    __syncthreads();

    const int fbase = ft * 16;
    const int n = nchunk * 128 + w * 16 + m;

    v8f acc_r, acc_i;
    complex_tile(shapes, w_tr, w_ti, s_inv, b, fbase, n, m, h, acc_r, acc_i);

    // epilogue: C/D element v -> (f = fbase + v + 8h, n)
    float rowsum[8];
    #pragma unroll
    for (int v = 0; v < 8; ++v) {
        const int fr = v + 8 * h;
        const float r = acc_r[v], im = acc_i[v];
        const float mag2 = r * r + im * im;
        const size_t bf = (size_t)(b * FF + fbase + fr);
        out_ln[bf * NWH + n] = (logf(mag2 + 1e-8f) + s_bias[fr]) * (1.0f / 4.2f);
        float s = mag2;                       // reduce across the 16 lanes of this half
        s += __shfl_down(s, 8, 16);
        s += __shfl_down(s, 4, 16);
        s += __shfl_down(s, 2, 16);
        s += __shfl_down(s, 1, 16);
        rowsum[v] = s;
    }
    if (m == 0) {
        #pragma unroll
        for (int v = 0; v < 8; ++v) s_rows[w][v + 8 * h] = rowsum[v];
    }
    __syncthreads();
    if (tid < 16) {
        float s = 0.f;
        #pragma unroll
        for (int ww = 0; ww < 8; ++ww) s += s_rows[ww][tid];
        w_fp[((size_t)(b * 8 + ft) * NCHUNKS + nchunk) * 16 + tid] = s;
    }
}

// 5) finalize 1/fnorm per (b,f)
__global__ __launch_bounds__(256) void k_fnorm(const float* __restrict__ w_fp,
                                               float* __restrict__ w_invf) {
    const int t = blockIdx.x * 256 + threadIdx.x;
    if (t >= BB * FF) return;
    const int b = t >> 7, f = t & 127, ft = f >> 4, r = f & 15;
    float s = 0.f;
    for (int c = 0; c < NCHUNKS; ++c)
        s += w_fp[((size_t)(b * 8 + ft) * NCHUNKS + c) * 16 + r];
    s = fmaxf(s, 1.1920929e-07f);     // FLT_EPSILON clip, then sqrt
    w_invf[t] = 1.0f / sqrtf(s);
}

// 6) GEMM pass 2: recompute tiles (shapes are L2-hot, math is ~free) and write
//    the normalized field exactly once. Same grid as pass 1.
__global__ __launch_bounds__(256) void k_gemm_write(const float* __restrict__ shapes,
                                                    const _Float16* __restrict__ w_tr,
                                                    const _Float16* __restrict__ w_ti,
                                                    const float* __restrict__ w_inv,
                                                    const float* __restrict__ w_invf,
                                                    float* __restrict__ out_field) {
    const int bx = blockIdx.x;
    const int nchunk = bx % NCHUNKS;
    const int ft = (bx / NCHUNKS) & 7;
    const int b  = bx / (NCHUNKS * 8);
    const int tid = threadIdx.x;
    const int w = tid >> 5;
    const int lane = tid & 31;
    const int m = lane & 15;
    const int h = lane >> 4;

    __shared__ float s_inv[32];
    __shared__ float s_invf[16];
    if (tid < 32) s_inv[tid] = w_inv[b * 32 + tid];
    if (tid < 16) s_invf[tid] = w_invf[b * FF + ft * 16 + tid];
    __syncthreads();

    const int fbase = ft * 16;
    const int n = nchunk * 128 + w * 16 + m;

    v8f acc_r, acc_i;
    complex_tile(shapes, w_tr, w_ti, s_inv, b, fbase, n, m, h, acc_r, acc_i);

    #pragma unroll
    for (int v = 0; v < 8; ++v) {
        const int fr = v + 8 * h;
        const float sc = s_invf[fr];
        const size_t bf = (size_t)(b * FF + fbase + fr);
        out_field[(bf * 2 + 0) * NWH + n] = acc_r[v] * sc;
        out_field[(bf * 2 + 1) * NWH + n] = acc_i[v] * sc;
    }
}

// 7) final aux scalar
__global__ __launch_bounds__(256) void k_auxfin(const float* __restrict__ w_aux,
                                                float* __restrict__ out_aux) {
    __shared__ float sm[8];
    float v = w_aux[threadIdx.x];
    v = block_reduce_256(v, sm);
    if (threadIdx.x == 0) *out_aux = v * (1.0f / 256.0f);
}

extern "C" void kernel_launch(void* const* d_in, const int* in_sizes, int n_in,
                              void* d_out, int out_size, void* d_ws, size_t ws_size,
                              hipStream_t stream) {
    const float* shapes = (const float*)d_in[0];   // (8,2,32,96,96)
    const float* resp   = (const float*)d_in[1];   // (8,128,2,32)
    const float* freqs  = (const float*)d_in[2];   // (8,128)
    const float* table  = (const float*)d_in[3];   // (250)

    float* out_ln    = (float*)d_out;              // (8,128,96,96)
    float* out_field = out_ln + LN_ELEMS;          // (8,128,2,96,96)
    float* out_aux   = out_ln + AUX_OFF;           // scalar

    float*    w_inv  = (float*)d_ws;               // 256
    float*    w_bias = w_inv + 256;                 // 1024
    float*    w_invf = w_bias + 1024;               // 1024
    float*    w_aux  = w_invf + 1024;               // 256
    float*    w_fp   = w_aux + 256;                 // 73728
    _Float16* w_tr   = (_Float16*)(w_fp + 73728);   // 32768 f16
    _Float16* w_ti   = w_tr + 32768;                // 32768 f16

    k_norm       <<<BB * PP,             256, 0, stream>>>(shapes, w_inv);
    k_prep       <<<(BB * FF + 255)/256, 256, 0, stream>>>(resp, freqs, table, w_bias, w_tr, w_ti);
    k_aux        <<<BB * PP,             256, 0, stream>>>(resp, w_aux);
    k_gemm_reduce<<<BB * 8 * NCHUNKS,    256, 0, stream>>>(shapes, w_tr, w_ti, w_inv, w_bias,
                                                           out_ln, w_fp);
    k_fnorm      <<<(BB * FF + 255)/256, 256, 0, stream>>>(w_fp, w_invf);
    k_gemm_write <<<BB * 8 * NCHUNKS,    256, 0, stream>>>(shapes, w_tr, w_ti, w_inv, w_invf,
                                                           out_field);
    k_auxfin     <<<1,                   256, 0, stream>>>(w_aux, out_aux);
}
